// ScaleAdaptiveRouter_9474697855375
// MI455X (gfx1250) — matrix operand order repaired and checked by reading, not compiled, not run on hardware
//
#include <hip/hip_runtime.h>
#include <hip/hip_bf16.h>
#include <stdint.h>

typedef __attribute__((ext_vector_type(16))) __bf16 v16bf;
typedef __attribute__((ext_vector_type(8)))  float  v8f;
typedef __attribute__((ext_vector_type(4)))  float  v4f;

static constexpr int kH    = 2048;   // hidden dim of x
static constexpr int kSE   = 64;     // scale-embedding dim
static constexpr int kKW   = 2112;   // router_w row length (H + SE)
static constexpr int kE    = 64;     // experts
static constexpr int kTok  = 16384;  // B * S
static constexpr int kWaves = 4;     // waves per block (128 threads)
static constexpr int kChunks = kH / 32;              // 64 K-chunks
static constexpr size_t kWsNeed = (size_t)kChunks * 4 /*nt*/ * 32 /*lanes*/ * 64; // 512 KB

// Convert 16 fp32 values into (hi, lo) bf16 fragments: f ~= hi + lo.
static __device__ __forceinline__ void cvt16(const float* __restrict__ f,
                                             v16bf& hi, v16bf& lo) {
#pragma unroll
  for (int i = 0; i < 16; ++i) {
    __bf16 h = (__bf16)f[i];
    hi[i] = h;
    lo[i] = (__bf16)(f[i] - (float)h);
  }
}

// ---------------------------------------------------------------------------
// Pre-pass: convert router_w[:, 0:2048] into WMMA B-fragment order, bf16 hi/lo.
// Layout: group g = chunk*4 + nt, per lane 16 dwords: [8 dwords hi][8 dwords lo].
// ---------------------------------------------------------------------------
__global__ __launch_bounds__(256) void preconv_w_kernel(
    const float* __restrict__ router_w, uint32_t* __restrict__ wsb) {
  const int g    = blockIdx.x * 256 + threadIdx.x;  // 0..8191 (256 groups * 32 lanes)
  const int lane = g & 31;
  const int grp  = g >> 5;       // chunk*4 + nt
  const int nt   = grp & 3;
  const int kk   = (grp >> 2) * 32;
  const int half = lane >> 4;
  const int l16  = lane & 15;

  const float* wr = router_w + (size_t)(nt * 16 + l16) * kKW + half * 16 + kk;
  float fb[16];
  *(v4f*)&fb[0]  = *(const v4f*)(wr);
  *(v4f*)&fb[4]  = *(const v4f*)(wr + 4);
  *(v4f*)&fb[8]  = *(const v4f*)(wr + 8);
  *(v4f*)&fb[12] = *(const v4f*)(wr + 12);
  v16bf hi, lo;
  cvt16(fb, hi, lo);

  uint32_t* dst = wsb + (size_t)g * 16;
  *(v16bf*)(dst)     = hi;
  *(v16bf*)(dst + 8) = lo;
}

// ---------------------------------------------------------------------------
// Main kernel: 1 M-tile (16 tokens) per wave, full 64 experts.
// ---------------------------------------------------------------------------
template <bool PRECONV>
__global__ __launch_bounds__(128) void scale_router_kernel(
    const float* __restrict__ x,            // [16384, 2048]
    const float* __restrict__ scale_emb,    // [10, 64]
    const float* __restrict__ router_w,     // [64, 2112]
    const float* __restrict__ noise,        // [16384, 64]
    const int*   __restrict__ scale_idx,    // [1]
    const uint32_t* __restrict__ wsb,       // preconverted B (if PRECONV)
    float*       __restrict__ out) {        // [16384, 64]
  __shared__ float s_logits[kWaves][16][kE];  // wave-private staging, 16KB
  __shared__ float s_bias[kE];

  const int tid  = threadIdx.x;
  const int wid  = tid >> 5;
  const int lane = tid & 31;
  const int half = lane >> 4;
  const int l16  = lane & 15;

  // Per-expert bias from the scale-embedding part of the concat (fp32 exact).
  if (wid == 0) {
    const float* emb = scale_emb + scale_idx[0] * kSE;
#pragma unroll
    for (int rep = 0; rep < 2; ++rep) {
      const int e = lane + rep * 32;
      const float* wr = router_w + (size_t)e * kKW + kH;
      float b = 0.f;
      for (int j = 0; j < kSE; ++j) b += emb[j] * wr[j];
      s_bias[e] = b;
    }
  }
  __syncthreads();

  const int wtile = blockIdx.x * kWaves + wid;  // 0..1023
  const int m0    = wtile * 16;

  v8f acc[4];
#pragma unroll
  for (int nt = 0; nt < 4; ++nt) acc[nt] = (v8f){};

  // A lane layout (16-bit, 16x32): row m = lane%16,
  // elems 0..15 = K(kk + half*16 + 0..15) contiguous.
  const float* xrow = x + (size_t)(m0 + l16) * kH + half * 16;

  for (int chunk = 0; chunk < kChunks; ++chunk) {
    const int kk = chunk * 32;
    float fa[16];
    *(v4f*)&fa[0]  = *(const v4f*)(xrow + kk);
    *(v4f*)&fa[4]  = *(const v4f*)(xrow + kk + 4);
    *(v4f*)&fa[8]  = *(const v4f*)(xrow + kk + 8);
    *(v4f*)&fa[12] = *(const v4f*)(xrow + kk + 12);
    v16bf ahi, alo;
    cvt16(fa, ahi, alo);

#pragma unroll
    for (int nt = 0; nt < 4; ++nt) {
      v16bf bhi, blo;
      if (PRECONV) {
        const uint32_t* wsl = wsb + ((size_t)(chunk * 4 + nt) * 32 + lane) * 16;
        bhi = *(const v16bf*)(wsl);
        blo = *(const v16bf*)(wsl + 8);
      } else {
        // B lane layout (16-bit, 32x16): column e = nt*16 + lane%16,
        // elems 0..15 = K(kk + half*16 + 0..15) contiguous.
        const float* wr = router_w + (size_t)(nt * 16 + l16) * kKW + half * 16 + kk;
        float fb[16];
        *(v4f*)&fb[0]  = *(const v4f*)(wr);
        *(v4f*)&fb[4]  = *(const v4f*)(wr + 4);
        *(v4f*)&fb[8]  = *(const v4f*)(wr + 8);
        *(v4f*)&fb[12] = *(const v4f*)(wr + 12);
        cvt16(fb, bhi, blo);
      }
      acc[nt] = __builtin_amdgcn_wmma_f32_16x16x32_bf16(
          false, ahi, false, bhi, (short)0, acc[nt], false, false);
      acc[nt] = __builtin_amdgcn_wmma_f32_16x16x32_bf16(
          false, alo, false, bhi, (short)0, acc[nt], false, false);
      acc[nt] = __builtin_amdgcn_wmma_f32_16x16x32_bf16(
          false, ahi, false, blo, (short)0, acc[nt], false, false);
    }
  }

  // Spill C to wave-private LDS (in-order DS per wave -> no barrier needed),
  // then lanes 0..15 each own one token for bias+noise + top-2 dispatch.
  // C layout: element r of v8f -> token m = r + half*8, expert e = nt*16 + lane%16.
#pragma unroll
  for (int nt = 0; nt < 4; ++nt)
#pragma unroll
    for (int r = 0; r < 8; ++r)
      s_logits[wid][r + half * 8][nt * 16 + l16] = acc[nt][r];

  if (lane < 16) {
    const int t = m0 + lane;
    const float* lrow = s_logits[wid][lane];
    const float* nrow = noise + (size_t)t * kE;
    float l1 = -3.4e38f, l2 = -3.4e38f;
    int e1 = 0, e2 = 0;
#pragma unroll 4
    for (int e = 0; e < kE; ++e) {
      float v = lrow[e] + s_bias[e] + 0.1f * nrow[e];
      if (v > l1)      { l2 = l1; e2 = e1; l1 = v; e1 = e; }
      else if (v > l2) { l2 = v; e2 = e; }
    }
    // normalized top-2 of softmax == sigmoid of logit gap (denominator cancels)
    const float rr = __expf(l2 - l1);
    const float w1 = 1.f / (1.f + rr);
    const float w2 = rr * w1;
    float* orow = out + (size_t)t * kE;
#pragma unroll
    for (int e = 0; e < kE; e += 4) {
      v4f o;
#pragma unroll
      for (int j = 0; j < 4; ++j) {
        const int ee = e + j;
        o[j] = (ee == e1) ? w1 : ((ee == e2) ? w2 : 0.f);
      }
      *(v4f*)(orow + e) = o;
    }
  }
}

extern "C" void kernel_launch(void* const* d_in, const int* in_sizes, int n_in,
                              void* d_out, int out_size, void* d_ws, size_t ws_size,
                              hipStream_t stream) {
  (void)in_sizes; (void)n_in; (void)out_size;
  const float* x    = (const float*)d_in[0];
  const float* se   = (const float*)d_in[1];
  const float* rw   = (const float*)d_in[2];
  const float* nz   = (const float*)d_in[3];
  const int*   sidx = (const int*)d_in[4];
  float* out = (float*)d_out;

  dim3 grid(kTok / (16 * kWaves));   // 256 blocks
  dim3 block(32 * kWaves);           // 128 threads (4 waves)

  if (ws_size >= kWsNeed && d_ws != nullptr) {
    uint32_t* wsb = (uint32_t*)d_ws;
    hipLaunchKernelGGL(preconv_w_kernel, dim3(32), dim3(256), 0, stream, rw, wsb);
    hipLaunchKernelGGL(scale_router_kernel<true>, grid, block, 0, stream,
                       x, se, rw, nz, sidx, wsb, out);
  } else {
    hipLaunchKernelGGL(scale_router_kernel<false>, grid, block, 0, stream,
                       x, se, rw, nz, sidx, nullptr, out);
  }
}
